// WaveletMoeGroupAttention_68135361184393
// MI455X (gfx1250) — matrix operand order repaired and checked by reading, not compiled, not run on hardware
//
#include <hip/hip_runtime.h>

#define HIDDEN 1024
#define QLEN   4096
#define BSZ    32
#define NHEADS 16
#define KVDIM  256

typedef __attribute__((ext_vector_type(16))) __bf16 v16bf;
typedef __attribute__((ext_vector_type(8)))  __bf16 v8bf;
typedef __attribute__((ext_vector_type(4)))  __bf16 v4bf;
typedef __attribute__((ext_vector_type(8)))  float  v8f;

__device__ __forceinline__ v8f wmma_bf16(v16bf a, v16bf b, v8f c) {
  return __builtin_amdgcn_wmma_f32_16x16x32_bf16(false, a, false, b, (short)0, c, false, false);
}

// XOR-swizzled LDS half-index; stride in halves (multiple of 8). 16B-chunk swizzle.
__device__ __forceinline__ int swi(int row, int colh, int stride) {
  int chunk = colh >> 3;
  return row * stride + (((chunk ^ (row & 15)) << 3) | (colh & 7));
}
// vs_t: stride 32 halves, only 4 chunks/row -> 2-bit xor
__device__ __forceinline__ int swiv(int row, int colh) {
  int chunk = colh >> 3;
  return row * 32 + (((chunk ^ (row & 3)) << 3) | (colh & 7));
}

// A operand (16x32 bf16): lane = row (mod 16), half-group hg = lane/16.
// elements 0..15 = K kbase+hg*16.. (16 contiguous K)
__device__ __forceinline__ v16bf ld_a(const __bf16* b, int row, int kbase, int hg, int stride) {
  union { v16bf v; v8bf p[2]; } u;
  u.p[0] = *(const v8bf*)(b + swi(row, kbase + hg * 16, stride));
  u.p[1] = *(const v8bf*)(b + swi(row, kbase + hg * 16 + 8, stride));
  return u.v;
}
__device__ __forceinline__ v16bf ld_a_plain(const __bf16* b, int row, int hg, int stride) {
  union { v16bf v; v8bf p[2]; } u;
  u.p[0] = *(const v8bf*)(b + row * stride + hg * 16);
  u.p[1] = *(const v8bf*)(b + row * stride + hg * 16 + 8);
  return u.v;
}
// B operand (32x16 bf16): lane = column (mod 16), 16 contiguous K at (lane/16)*16
__device__ __forceinline__ v16bf ld_b_lds(const __bf16* b, int row, int colh, int stride) {
  union { v16bf v; v8bf p[2]; } u;
  u.p[0] = *(const v8bf*)(b + swi(row, colh, stride));
  u.p[1] = *(const v8bf*)(b + swi(row, colh + 8, stride));
  return u.v;
}
__device__ __forceinline__ v16bf ld_b_vst(const __bf16* b, int row, int colh) {
  union { v16bf v; v8bf p[2]; } u;
  u.p[0] = *(const v8bf*)(b + swiv(row, colh));
  u.p[1] = *(const v8bf*)(b + swiv(row, colh + 8));
  return u.v;
}
__device__ __forceinline__ v16bf ld_b_glb(const __bf16* p) {
  union { v16bf v; v8bf p2[2]; } u;
  u.p2[0] = *(const v8bf*)(p);
  u.p2[1] = *(const v8bf*)(p + 8);
  return u.v;
}

__global__ void cvt_f32_bf16(const float* __restrict__ src, unsigned short* __restrict__ dst, int n4) {
  int i = blockIdx.x * 256 + threadIdx.x;
  if (i < n4) {
    float4 f = ((const float4*)src)[i];
    v4bf h;
    h[0] = (__bf16)f.x; h[1] = (__bf16)f.y; h[2] = (__bf16)f.z; h[3] = (__bf16)f.w;
    *(v4bf*)(dst + i * 4) = h;
  }
}

__global__ __launch_bounds__(256) void fused_gqa_attn(
    const float* __restrict__ x, const float* __restrict__ mask,
    const float* __restrict__ bq, const float* __restrict__ bk, const float* __restrict__ bv,
    const unsigned short* __restrict__ wq_u, const unsigned short* __restrict__ wk_u,
    const unsigned short* __restrict__ wv_u, const unsigned short* __restrict__ wo_u,
    float* __restrict__ out) {
  __shared__ __align__(16) unsigned short smem[81920];  // 160 KB exactly
  __bf16* xs  = (__bf16*)smem;        // [32][1024] swizzled  (x, later softmax scratch)
  __bf16* qs  = xs + 32 * 1024;       // [32][1024] swizzled  (Q, later attn out)
  __bf16* ksm = qs + 32 * 1024;       // [32][256]  swizzled  (K: batch x kvdim)
  __bf16* vst = ksm + 32 * 256;       // [256][32]  swizzled  (V^T: kvdim x batch)

  const __bf16* wq = (const __bf16*)wq_u;
  const __bf16* wk = (const __bf16*)wk_u;
  const __bf16* wv = (const __bf16*)wv_u;
  const __bf16* wo = (const __bf16*)wo_u;

  const int t = threadIdx.x;
  const int w = t >> 5, l = t & 31;
  const int ln = l & 15, hg = l >> 4;
  const int q = blockIdx.x;
  const v8f vzero = {0.f, 0.f, 0.f, 0.f, 0.f, 0.f, 0.f, 0.f};

  // ---- Phase 0: x[., q, .] -> xs (bf16, swizzled) ----
  for (int rr = 0; rr < 4; ++rr) {
    int row = w * 4 + rr;
    const float4* src = (const float4*)(x + (size_t)row * QLEN * HIDDEN + (size_t)q * HIDDEN);
    #pragma unroll
    for (int it = 0; it < 8; ++it) {
      float4 f = src[it * 32 + l];
      int d0 = (it * 32 + l) * 4;
      v4bf h;
      h[0] = (__bf16)f.x; h[1] = (__bf16)f.y; h[2] = (__bf16)f.z; h[3] = (__bf16)f.w;
      *(v4bf*)(xs + swi(row, d0, 1024)) = h;
    }
  }
  __syncthreads();

  // ---- Phase 1a: Q projection (wave w -> ntiles 8w..8w+7, two passes of 4) ----
  for (int pass = 0; pass < 2; ++pass) {
    int nb = w * 8 + pass * 4;
    v8f acc[2][4];
    #pragma unroll
    for (int mt = 0; mt < 2; ++mt)
      #pragma unroll
      for (int j = 0; j < 4; ++j) acc[mt][j] = vzero;
    for (int k = 0; k < 32; ++k) {
      int kb = k * 32;
      v16bf a0 = ld_a(xs, ln, kb, hg, 1024);
      v16bf a1 = ld_a(xs, 16 + ln, kb, hg, 1024);
      #pragma unroll
      for (int j = 0; j < 4; ++j) {
        v16bf b = ld_b_glb(wq + (size_t)((nb + j) * 16 + ln) * 1024 + kb + hg * 16);
        acc[0][j] = wmma_bf16(a0, b, acc[0][j]);
        acc[1][j] = wmma_bf16(a1, b, acc[1][j]);
      }
    }
    #pragma unroll
    for (int j = 0; j < 4; ++j) {
      int col = (nb + j) * 16 + ln;
      float bias = bq[col];
      #pragma unroll
      for (int mt = 0; mt < 2; ++mt)
        #pragma unroll
        for (int r = 0; r < 8; ++r) {
          int row = mt * 16 + r + 8 * hg;
          qs[swi(row, col, 1024)] = (__bf16)(acc[mt][j][r] + bias);
        }
    }
  }

  // ---- Phase 1b: K projection (wave w -> ntiles 2w, 2w+1) ----
  {
    v8f acc[2][2];
    #pragma unroll
    for (int mt = 0; mt < 2; ++mt) { acc[mt][0] = vzero; acc[mt][1] = vzero; }
    for (int k = 0; k < 32; ++k) {
      int kb = k * 32;
      v16bf a0 = ld_a(xs, ln, kb, hg, 1024);
      v16bf a1 = ld_a(xs, 16 + ln, kb, hg, 1024);
      #pragma unroll
      for (int j = 0; j < 2; ++j) {
        v16bf b = ld_b_glb(wk + (size_t)((w * 2 + j) * 16 + ln) * 1024 + kb + hg * 16);
        acc[0][j] = wmma_bf16(a0, b, acc[0][j]);
        acc[1][j] = wmma_bf16(a1, b, acc[1][j]);
      }
    }
    #pragma unroll
    for (int j = 0; j < 2; ++j) {
      int col = (w * 2 + j) * 16 + ln;
      float bias = bk[col];
      #pragma unroll
      for (int mt = 0; mt < 2; ++mt)
        #pragma unroll
        for (int r = 0; r < 8; ++r) {
          int row = mt * 16 + r + 8 * hg;
          ksm[swi(row, col, 256)] = (__bf16)(acc[mt][j][r] + bias);
        }
    }
  }

  // ---- Phase 1c: V projection -> vst (transposed) ----
  {
    v8f acc[2][2];
    #pragma unroll
    for (int mt = 0; mt < 2; ++mt) { acc[mt][0] = vzero; acc[mt][1] = vzero; }
    for (int k = 0; k < 32; ++k) {
      int kb = k * 32;
      v16bf a0 = ld_a(xs, ln, kb, hg, 1024);
      v16bf a1 = ld_a(xs, 16 + ln, kb, hg, 1024);
      #pragma unroll
      for (int j = 0; j < 2; ++j) {
        v16bf b = ld_b_glb(wv + (size_t)((w * 2 + j) * 16 + ln) * 1024 + kb + hg * 16);
        acc[0][j] = wmma_bf16(a0, b, acc[0][j]);
        acc[1][j] = wmma_bf16(a1, b, acc[1][j]);
      }
    }
    #pragma unroll
    for (int j = 0; j < 2; ++j) {
      int col = (w * 2 + j) * 16 + ln;  // kv-dim index (row of vst)
      float bias = bv[col];
      #pragma unroll
      for (int mt = 0; mt < 2; ++mt)
        #pragma unroll
        for (int r = 0; r < 8; ++r) {
          int row = mt * 16 + r + 8 * hg;  // batch index (col of vst)
          vst[swiv(col, row)] = (__bf16)(acc[mt][j][r] + bias);
        }
    }
  }
  __syncthreads();

  // ---- Phase 2: attention, 2 heads per wave ----
  float* sc = (float*)smem + w * 2048;     // per-wave scratch in xs region: 32x33 f32
  __bf16* pb = (__bf16*)(sc + 32 * 33);    // 32x40 bf16 probs (A-layout friendly)
  const float scale = 0.125f;              // 1/sqrt(64)
  const float* mrow = mask + (size_t)q * (BSZ * BSZ);

  for (int hh = 0; hh < 2; ++hh) {
    int head = w * 2 + hh;
    int kvb = (head >> 2) * 64;  // GQA: head h uses kv-head h/4

    // scores = Q Kt : A rows=batch from qs cols head*64.., B cols=batch from ksm
    v8f s[2][2];
    #pragma unroll
    for (int mt = 0; mt < 2; ++mt) { s[mt][0] = vzero; s[mt][1] = vzero; }
    #pragma unroll
    for (int ks2 = 0; ks2 < 2; ++ks2) {
      int kbq = head * 64 + ks2 * 32;
      int kbk = kvb + ks2 * 32;
      v16bf a0 = ld_a(qs, ln, kbq, hg, 1024);
      v16bf a1 = ld_a(qs, 16 + ln, kbq, hg, 1024);
      #pragma unroll
      for (int ct = 0; ct < 2; ++ct) {
        v16bf b = ld_b_lds(ksm, ct * 16 + ln, kbk + hg * 16, 256);
        s[0][ct] = wmma_bf16(a0, b, s[0][ct]);
        s[1][ct] = wmma_bf16(a1, b, s[1][ct]);
      }
    }
    #pragma unroll
    for (int mt = 0; mt < 2; ++mt)
      #pragma unroll
      for (int ct = 0; ct < 2; ++ct)
        #pragma unroll
        for (int r = 0; r < 8; ++r) {
          int row = mt * 16 + r + 8 * hg;
          int col = ct * 16 + ln;
          sc[row * 33 + col] = s[mt][ct][r] * scale + mrow[row * 32 + col];
        }
    __syncthreads();

    // softmax: lane l owns row l (32 lanes == 32 batch rows)
    {
      float mx = -3.0e38f;
      for (int c = 0; c < 32; ++c) mx = fmaxf(mx, sc[l * 33 + c]);
      float sum = 0.f;
      for (int c = 0; c < 32; ++c) {
        float e = __expf(sc[l * 33 + c] - mx);
        sc[l * 33 + c] = e;
        sum += e;
      }
      float inv = 1.f / sum;
      for (int c = 0; c < 32; ++c) pb[l * 40 + c] = (__bf16)(sc[l * 33 + c] * inv);
    }
    __syncthreads();

    // attn = probs @ V : B from vst (column of V is a contiguous row of V^T)
    v8f o[2][4];
    #pragma unroll
    for (int mt = 0; mt < 2; ++mt)
      #pragma unroll
      for (int nt = 0; nt < 4; ++nt) o[mt][nt] = vzero;
    {
      v16bf a0 = ld_a_plain(pb, ln, hg, 40);
      v16bf a1 = ld_a_plain(pb, 16 + ln, hg, 40);
      #pragma unroll
      for (int nt = 0; nt < 4; ++nt) {
        v16bf b = ld_b_vst(vst, kvb + nt * 16 + ln, hg * 16);
        o[0][nt] = wmma_bf16(a0, b, o[0][nt]);
        o[1][nt] = wmma_bf16(a1, b, o[1][nt]);
      }
    }
    // store attn into qs (overwrites this head's now-dead Q columns)
    #pragma unroll
    for (int nt = 0; nt < 4; ++nt)
      #pragma unroll
      for (int mt = 0; mt < 2; ++mt)
        #pragma unroll
        for (int r = 0; r < 8; ++r) {
          int row = mt * 16 + r + 8 * hg;
          int col = head * 64 + nt * 16 + ln;
          qs[swi(row, col, 1024)] = (__bf16)o[mt][nt][r];
        }
    __syncthreads();
  }

  // ---- Phase 3: output projection attn @ Wo^T -> out[b, q, :] ----
  for (int pass = 0; pass < 2; ++pass) {
    int nb = w * 8 + pass * 4;
    v8f acc[2][4];
    #pragma unroll
    for (int mt = 0; mt < 2; ++mt)
      #pragma unroll
      for (int j = 0; j < 4; ++j) acc[mt][j] = vzero;
    for (int k = 0; k < 32; ++k) {
      int kb = k * 32;
      v16bf a0 = ld_a(qs, ln, kb, hg, 1024);
      v16bf a1 = ld_a(qs, 16 + ln, kb, hg, 1024);
      #pragma unroll
      for (int j = 0; j < 4; ++j) {
        v16bf b = ld_b_glb(wo + (size_t)((nb + j) * 16 + ln) * 1024 + kb + hg * 16);
        acc[0][j] = wmma_bf16(a0, b, acc[0][j]);
        acc[1][j] = wmma_bf16(a1, b, acc[1][j]);
      }
    }
    #pragma unroll
    for (int j = 0; j < 4; ++j) {
      int col = (nb + j) * 16 + ln;
      #pragma unroll
      for (int mt = 0; mt < 2; ++mt)
        #pragma unroll
        for (int r = 0; r < 8; ++r) {
          int row = mt * 16 + r + 8 * hg;
          out[(size_t)row * QLEN * HIDDEN + (size_t)q * HIDDEN + col] = acc[mt][j][r];
        }
    }
  }
}

extern "C" void kernel_launch(void* const* d_in, const int* in_sizes, int n_in,
                              void* d_out, int out_size, void* d_ws, size_t ws_size,
                              hipStream_t stream) {
  const float* x    = (const float*)d_in[0];
  const float* mask = (const float*)d_in[1];
  const float* Wq   = (const float*)d_in[2];
  const float* bq   = (const float*)d_in[3];
  const float* Wk   = (const float*)d_in[4];
  const float* bk   = (const float*)d_in[5];
  const float* Wv   = (const float*)d_in[6];
  const float* bv   = (const float*)d_in[7];
  const float* Wo   = (const float*)d_in[8];

  unsigned short* wq_bf = (unsigned short*)d_ws;
  unsigned short* wk_bf = wq_bf + 1024 * 1024;
  unsigned short* wv_bf = wk_bf + 256 * 1024;
  unsigned short* wo_bf = wv_bf + 256 * 1024;

  // bf16 weight conversion (5 MB total -> L2-resident for the fused kernel)
  cvt_f32_bf16<<<(1024 * 1024 / 4) / 256, 256, 0, stream>>>(Wq, wq_bf, 1024 * 1024 / 4);
  cvt_f32_bf16<<<(256 * 1024 / 4) / 256, 256, 0, stream>>>(Wk, wk_bf, 256 * 1024 / 4);
  cvt_f32_bf16<<<(256 * 1024 / 4) / 256, 256, 0, stream>>>(Wv, wv_bf, 256 * 1024 / 4);
  cvt_f32_bf16<<<(1024 * 1024 / 4) / 256, 256, 0, stream>>>(Wo, wo_bf, 1024 * 1024 / 4);

  fused_gqa_attn<<<QLEN, 256, 0, stream>>>(x, mask, bq, bk, bv,
                                           wq_bf, wk_bf, wv_bf, wo_bf, (float*)d_out);
}
